// GNNBranchE3_6725918786006
// MI455X (gfx1250) — compile-verified, matrix operand order checked by reading, not run on hardware
//
#include <hip/hip_runtime.h>
#include <math.h>

#define NND 30000
#define NED 320000
#define NGR 16
#define ESMD 1280
#define LIGD 43
#define EPSF 1e-5f
#define Y0C 0.28209479177387814f
#define Y1C 0.4886025119029199f

typedef _Float16 h16;
typedef __attribute__((ext_vector_type(16))) _Float16 v16h;
typedef __attribute__((ext_vector_type(8)))  float    v8f;

// f16 weights pre-swizzled to WMMA fragment order + f32 bias
struct TPW { const h16 *Wss, *Wsv, *Wvs, *Wvv_s, *Wvv_v; const float* b; };

// ---------------------------------------------------------------------------
// WMMA 16x16x32 f16 fragment order: element (m,k) of a 16xK tile lives at
//   [k>>5][hh*16+m][slot],  hh = (k>>3)&1,  slot = ((k>>4)<<3)|(k&7)
// so each lane's 16 halves (one v16h) are contiguous 32 bytes.
__device__ __forceinline__ int fragidx(int m, int k) {
  int kt = k >> 5, kk = k & 31;
  int hh = (kk >> 3) & 1;
  int slot = ((kk >> 4) << 3) | (kk & 7);
  return kt * 512 + (hh * 16 + m) * 16 + slot;
}

__device__ __forceinline__ v8f wmma32(v16h a, v16h b, v8f c) {
  return __builtin_amdgcn_wmma_f32_16x16x32_f16(false, a, false, b, (short)0, c,
                                                false, false);
}

__device__ __forceinline__ v16h ldfragB(const h16* __restrict__ B, int kt, int ntiles,
                                        int nt, int lane) {
  return *(const v16h*)(B + ((size_t)kt * ntiles + nt) * 512 + lane * 16);
}

// ---------------- Generic lmax=1 fully-connected tensor product --------------
// 16-row tile shared by NW waves; n-tiles distributed wave-cyclically so each
// wave's WMMAs run with full EXEC.  A fragments cached in VGPRs.
// Dso [16][SOUT] = (s*a0)@Wss + b + (v.a1)@Wvv_s
// Vout [16][VOUT*3] = (s@Wsv)(x)a1 + (v*a0)@Wvs + (v x a1)@Wvv_v
template <int NW, int KTS, int VIN, int SOUT, int VOUT>
__device__ void tp_eval(const TPW w,
                        const h16* __restrict__ SinF, const h16* __restrict__ SinSF,
                        const h16* __restrict__ VinN,
                        const float* __restrict__ a0, const float* __restrict__ a1,
                        float* Dso, float* Dsv, float* Vout, h16* tA, h16* tA2) {
  constexpr int NTHR = NW * 32;
  constexpr int KTV = (VIN > 0) ? VIN / 32 : 1;
  constexpr int NTS = SOUT / 16, NTV = VOUT / 16;
  const int tid = threadIdx.x;
  const int lane = tid & 31, wid = tid >> 5;
  const int n = lane & 15, mb = (lane >> 4) * 8;
  if constexpr (VIN > 0) {  // vdot = v . a1 (unscaled), frag order
    for (int t = tid; t < 16 * VIN; t += NTHR) {
      int m = t / VIN, i = t % VIN;
      const h16* vp = VinN + m * VIN * 3 + i * 3;
      float d = (float)vp[0] * a1[m * 3 + 0] + (float)vp[1] * a1[m * 3 + 1] +
                (float)vp[2] * a1[m * 3 + 2];
      tA[fragidx(m, i)] = (h16)d;
    }
  }
  __syncthreads();
  {
    v16h aS[KTS];
#pragma unroll
    for (int kt = 0; kt < KTS; ++kt)
      aS[kt] = *(const v16h*)(SinSF + kt * 512 + lane * 16);
    v16h aD[KTV];
    if constexpr (VIN > 0) {
#pragma unroll
      for (int kt = 0; kt < KTV; ++kt)
        aD[kt] = *(const v16h*)(tA + kt * 512 + lane * 16);
    }
    for (int nt = wid; nt < NTS; nt += NW) {
      v8f acc = {};
#pragma unroll
      for (int kt = 0; kt < KTS; ++kt)
        acc = wmma32(aS[kt], ldfragB(w.Wss, kt, NTS, nt, lane), acc);
      if constexpr (VIN > 0) {
#pragma unroll
        for (int kt = 0; kt < KTV; ++kt)
          acc = wmma32(aD[kt], ldfragB(w.Wvv_s, kt, NTS, nt, lane), acc);
      }
      float bias = w.b[nt * 16 + n];
#pragma unroll
      for (int r = 0; r < 8; ++r) Dso[(mb + r) * SOUT + nt * 16 + n] = acc[r] + bias;
    }
    v16h aU[KTS];
#pragma unroll
    for (int kt = 0; kt < KTS; ++kt)
      aU[kt] = *(const v16h*)(SinF + kt * 512 + lane * 16);
    for (int nt = wid; nt < NTV; nt += NW) {
      v8f acc = {};
#pragma unroll
      for (int kt = 0; kt < KTS; ++kt)
        acc = wmma32(aU[kt], ldfragB(w.Wsv, kt, NTV, nt, lane), acc);
#pragma unroll
      for (int r = 0; r < 8; ++r) Dsv[(mb + r) * VOUT + nt * 16 + n] = acc[r];
    }
  }
  __syncthreads();
#pragma unroll
  for (int c = 0; c < 3; ++c) {
    if constexpr (VIN > 0) {
      const int c1 = (c + 1) % 3, c2 = (c + 2) % 3;
      for (int t = tid; t < 16 * VIN; t += NTHR) {
        int m = t / VIN, i = t % VIN;
        const h16* vp = VinN + m * VIN * 3 + i * 3;
        int fi = fragidx(m, i);
        tA[fi]  = (h16)((float)vp[c] * a0[m]);
        tA2[fi] = (h16)((float)vp[c1] * a1[m * 3 + c2] -
                        (float)vp[c2] * a1[m * 3 + c1]);
      }
      __syncthreads();
      v16h aV[KTV], aX[KTV];
#pragma unroll
      for (int kt = 0; kt < KTV; ++kt) {
        aV[kt] = *(const v16h*)(tA + kt * 512 + lane * 16);
        aX[kt] = *(const v16h*)(tA2 + kt * 512 + lane * 16);
      }
      for (int nt = wid; nt < NTV; nt += NW) {
        v8f acc = {};
#pragma unroll
        for (int kt = 0; kt < KTV; ++kt)
          acc = wmma32(aV[kt], ldfragB(w.Wvs, kt, NTV, nt, lane), acc);
#pragma unroll
        for (int kt = 0; kt < KTV; ++kt)
          acc = wmma32(aX[kt], ldfragB(w.Wvv_v, kt, NTV, nt, lane), acc);
#pragma unroll
        for (int r = 0; r < 8; ++r) {
          int m = mb + r, oo = nt * 16 + n;
          Vout[(m * VOUT + oo) * 3 + c] = acc[r] + Dsv[m * VOUT + oo] * a1[m * 3 + c];
        }
      }
      __syncthreads();
    } else {
      for (int t = tid; t < 16 * VOUT; t += NTHR) {
        int m = t / VOUT;
        Vout[t * 3 + c] = Dsv[t] * a1[m * 3 + c];
      }
    }
  }
  __syncthreads();
}

// SwishGate: Sout = silu(Dso[:, :h]); Vout *= sigmoid(Dso[:, h:2h])
__device__ void tp_gate(int h, const float* Dso, int sout, float* Sout, float* Vout) {
  for (int t = threadIdx.x; t < 16 * h; t += blockDim.x) {
    int m = t / h, j = t % h;
    float xv = Dso[m * sout + j];
    Sout[t] = xv / (1.f + __expf(-xv));
    float g = Dso[m * sout + h + j];
    g = 1.f / (1.f + __expf(-g));
    Vout[t * 3 + 0] *= g; Vout[t * 3 + 1] *= g; Vout[t * 3 + 2] *= g;
  }
}

// ---------------- Utility kernels --------------------------------------------
__global__ void k_fill(float* __restrict__ p, float v, int n) {
  int i = blockIdx.x * blockDim.x + threadIdx.x;
  if (i < n) p[i] = v;
}

// convert f32 weight [K][N] -> f16 frag-tile order [kt][nt][32][16], zero-pad K
__global__ void k_cvt_w(const float* __restrict__ W, h16* __restrict__ out,
                        int K, int N, int Kpad) {
  int t = blockIdx.x * blockDim.x + threadIdx.x;
  if (t >= Kpad * N) return;
  int k = t / N, n = t % N;
  float val = (k < K) ? W[(size_t)k * N + n] : 0.f;
  int kt = k >> 5, kk = k & 31, nt = n >> 4, nn = n & 15;
  int hh = (kk >> 3) & 1;
  int slot = ((kk >> 4) << 3) | (kk & 7);
  int ntiles = N >> 4;
  out[((size_t)kt * ntiles + nt) * 512 + (hh * 16 + nn) * 16 + slot] = (h16)val;
}

__global__ void k_counts(const int* __restrict__ batch, float* __restrict__ counts, int n) {
  int i = blockIdx.x * blockDim.x + threadIdx.x;
  if (i < n) atomicAdd(&counts[batch[i]], 1.f);
}

// ---------------- Input projection (WMMA 30000x1280x64) ----------------------
__global__ __launch_bounds__(32) void k_inproj(
    const float* __restrict__ x, const int* __restrict__ ntype,
    const h16* __restrict__ protWf, const float* __restrict__ protB,
    const float* __restrict__ protG, const float* __restrict__ protLB,
    const float* __restrict__ ligW, const float* __restrict__ ligB,
    const float* __restrict__ ligG, const float* __restrict__ ligLB,
    const float* __restrict__ nodeEmb, float* __restrict__ s) {
  __shared__ __align__(32) h16 Ach[512];
  __shared__ float Dp[1024];
  __shared__ float Dl[1024];
  const int lane = threadIdx.x;
  const int r0 = blockIdx.x * 16;
  v8f acc[4];
#pragma unroll
  for (int i = 0; i < 4; ++i) acc[i] = (v8f){};
  for (int k0 = 0; k0 < ESMD; k0 += 32) {
    for (int t = lane; t < 512; t += 32) {
      int m = t >> 5, kk = t & 31;
      Ach[fragidx(m, kk)] = (h16)x[(size_t)(r0 + m) * ESMD + k0 + kk];
    }
    __syncthreads();
    v16h af = *(const v16h*)(Ach + lane * 16);
#pragma unroll
    for (int nt = 0; nt < 4; ++nt) {
      v16h bf = *(const v16h*)(protWf + ((size_t)(k0 >> 5) * 4 + nt) * 512 + lane * 16);
      acc[nt] = wmma32(af, bf, acc[nt]);
    }
    __syncthreads();
  }
  {
    int n = lane & 15, mb = (lane >> 4) * 8;
    for (int nt = 0; nt < 4; ++nt)
#pragma unroll
      for (int r = 0; r < 8; ++r) Dp[(mb + r) * 64 + nt * 16 + n] = acc[nt][r];
  }
  __syncthreads();
  for (int t = lane; t < 1024; t += 32) {
    int m = t >> 6, j = t & 63;
    Dp[t] += protB[j];
    float a = ligB[j];
    const float* xr = x + (size_t)(r0 + m) * ESMD;
    for (int i = 0; i < LIGD; ++i) a += xr[i] * ligW[i * 64 + j];
    Dl[t] = a;
  }
  __syncthreads();
  if (lane < 16) {
    const int m = lane;
    float mp = 0.f, ml = 0.f;
    for (int j = 0; j < 64; ++j) { mp += Dp[m * 64 + j]; ml += Dl[m * 64 + j]; }
    mp /= 64.f; ml /= 64.f;
    float vp = 0.f, vl = 0.f;
    for (int j = 0; j < 64; ++j) {
      float d1 = Dp[m * 64 + j] - mp; vp += d1 * d1;
      float d2 = Dl[m * 64 + j] - ml; vl += d2 * d2;
    }
    float rp = rsqrtf(vp / 64.f + EPSF), rl = rsqrtf(vl / 64.f + EPSF);
    int ty = ntype[r0 + m];
    for (int j = 0; j < 64; ++j) {
      float pv = fmaxf((Dp[m * 64 + j] - mp) * rp * protG[j] + protLB[j], 0.f);
      float lv = fmaxf((Dl[m * 64 + j] - ml) * rl * ligG[j] + ligLB[j], 0.f);
      float o = (ty == 0 ? pv : 0.f) + (ty == 1 ? lv : 0.f) + nodeEmb[ty * 64 + j];
      s[(size_t)(r0 + m) * 64 + j] = o;
    }
  }
}

// ---------------- Edge geometry + node attr accumulation ---------------------
__global__ void k_edge_geom(const float* __restrict__ pos, const int* __restrict__ eidx,
                            float* __restrict__ dist, float* __restrict__ ea1,
                            float* __restrict__ deg, float* __restrict__ na1, int E) {
  int e = blockIdx.x * blockDim.x + threadIdx.x;
  if (e >= E) return;
  int si = eidx[e], di = eidx[E + e];
  float rx = pos[si * 3 + 0] - pos[di * 3 + 0];
  float ry = pos[si * 3 + 1] - pos[di * 3 + 1];
  float rz = pos[si * 3 + 2] - pos[di * 3 + 2];
  float d = sqrtf(rx * rx + ry * ry + rz * rz);
  float inv = 1.f / fmaxf(d, 1e-12f);
  float ex = Y1C * rx * inv, ey = Y1C * ry * inv, ez = Y1C * rz * inv;
  dist[e] = d;
  ea1[e * 3 + 0] = ex; ea1[e * 3 + 1] = ey; ea1[e * 3 + 2] = ez;
  atomicAdd(&deg[di], 1.f);
  atomicAdd(&na1[di * 3 + 0], ex);
  atomicAdd(&na1[di * 3 + 1], ey);
  atomicAdd(&na1[di * 3 + 2], ez);
}

__global__ void k_fin_na(const float* __restrict__ deg, float* __restrict__ na0,
                         float* __restrict__ na1, int n) {
  int i = blockIdx.x * blockDim.x + threadIdx.x;
  if (i >= n) return;
  float dg = deg[i], dc = fmaxf(dg, 1.f);
  na0[i] = Y0C * dg / dc;
  na1[i * 3 + 0] /= dc; na1[i * 3 + 1] /= dc; na1[i * 3 + 2] /= dc;
}

// ---------------- Embedding TP (scalar-only input) ---------------------------
__global__ __launch_bounds__(32) void k_embed(float* __restrict__ s, float* __restrict__ v,
                                              const float* __restrict__ na0,
                                              const float* __restrict__ na1, TPW w) {
  __shared__ __align__(32) h16 SinF[2 * 512];
  __shared__ __align__(32) h16 SinSF[2 * 512];
  __shared__ float Dso[16 * 64];
  __shared__ float Dsv[16 * 64];
  __shared__ float Vout[16 * 64 * 3];
  __shared__ float a0s[16], a1s[48];
  const int lane = threadIdx.x;
  const int n0 = blockIdx.x * 16;
  if (lane < 16) {
    a0s[lane] = na0[n0 + lane];
    a1s[lane * 3 + 0] = na1[(n0 + lane) * 3 + 0];
    a1s[lane * 3 + 1] = na1[(n0 + lane) * 3 + 1];
    a1s[lane * 3 + 2] = na1[(n0 + lane) * 3 + 2];
  }
  __syncthreads();
  for (int t = lane; t < 1024; t += 32) {
    int m = t >> 6, j = t & 63;
    float val = s[(size_t)(n0 + m) * 64 + j];
    int fi = fragidx(m, j);
    SinF[fi] = (h16)val;
    SinSF[fi] = (h16)(val * a0s[m]);
  }
  __syncthreads();
  tp_eval<1, 2, 0, 64, 64>(w, SinF, SinSF, nullptr, a0s, a1s, Dso, Dsv, Vout,
                           nullptr, nullptr);
  for (int t = lane; t < 1024; t += 32) {
    int m = t >> 6, j = t & 63;
    s[(size_t)(n0 + m) * 64 + j] = Dso[t];
    v[(size_t)(n0 + m) * 192 + j * 3 + 0] = Vout[t * 3 + 0];
    v[(size_t)(n0 + m) * 192 + j * 3 + 1] = Vout[t * 3 + 1];
    v[(size_t)(n0 + m) * 192 + j * 3 + 2] = Vout[t * 3 + 2];
  }
}

// ---------- fused msg1 -> gate -> msg2 -> gate -> atomic scatter -------------
// 4 waves cooperate on one 16-edge tile.
__global__ __launch_bounds__(128) void k_edge_msg(
    const float* __restrict__ s, const float* __restrict__ v,
    const float* __restrict__ dist, const float* __restrict__ ea1,
    const float* __restrict__ edge_emb, const int* __restrict__ eidx,
    const int* __restrict__ etype, TPW w1, TPW w2,
    float* __restrict__ msagg, float* __restrict__ mvagg, int E) {
  __shared__ __align__(32) h16 SinF[5 * 512];
  __shared__ __align__(32) h16 SinSF[5 * 512];
  __shared__ h16 VinN[16 * 128 * 3];
  __shared__ __align__(32) h16 tA[4 * 512];
  __shared__ __align__(32) h16 tA2[4 * 512];
  __shared__ float Dso[16 * 128];
  __shared__ float Dsv[16 * 64];
  __shared__ float Vout[16 * 64 * 3];
  __shared__ float Sout[16 * 64];
  __shared__ float a0s[16], a1s[48];
  __shared__ int nsrc[16], ndst[16];
  const int tid = threadIdx.x;
  const int e0 = blockIdx.x * 16;
  if (tid < 16) {
    int e = e0 + tid;
    nsrc[tid] = eidx[e];
    ndst[tid] = eidx[E + e];
    a0s[tid] = Y0C;
    a1s[tid * 3 + 0] = ea1[e * 3 + 0];
    a1s[tid * 3 + 1] = ea1[e * 3 + 1];
    a1s[tid * 3 + 2] = ea1[e * 3 + 2];
  }
  __syncthreads();
  for (int t = tid; t < 16 * 160; t += 128) {
    int m = t / 160, j = t % 160;
    float val;
    if (j < 64)        val = s[(size_t)ndst[m] * 64 + j];
    else if (j < 128)  val = s[(size_t)nsrc[m] * 64 + (j - 64)];
    else if (j == 128) val = dist[e0 + m];
    else if (j < 145)  val = edge_emb[etype[e0 + m] * 16 + (j - 129)];
    else               val = 0.f;
    int fi = fragidx(m, j);
    SinF[fi] = (h16)val;
    SinSF[fi] = (h16)(val * Y0C);
  }
  for (int t = tid; t < 16 * 128 * 3; t += 128) {
    int m = t / 384, r = t % 384, i = r / 3, c = r % 3;
    float val = (i < 64) ? v[(size_t)ndst[m] * 192 + i * 3 + c]
                         : v[(size_t)nsrc[m] * 192 + (i - 64) * 3 + c];
    VinN[t] = (h16)val;
  }
  __syncthreads();
  tp_eval<4, 5, 128, 128, 64>(w1, SinF, SinSF, VinN, a0s, a1s, Dso, Dsv, Vout, tA, tA2);
  tp_gate(64, Dso, 128, Sout, Vout);
  __syncthreads();
  for (int t = tid; t < 1024; t += 128) {
    int m = t >> 6, j = t & 63;
    float val = Sout[t];
    int fi = fragidx(m, j);
    SinF[fi] = (h16)val;
    SinSF[fi] = (h16)(val * Y0C);
  }
  for (int t = tid; t < 3072; t += 128) VinN[t] = (h16)Vout[t];
  __syncthreads();
  tp_eval<4, 2, 64, 128, 64>(w2, SinF, SinSF, VinN, a0s, a1s, Dso, Dsv, Vout, tA, tA2);
  tp_gate(64, Dso, 128, Sout, Vout);
  __syncthreads();
  for (int t = tid; t < 1024; t += 128) {
    int m = t >> 6, j = t & 63;
    atomicAdd(&msagg[(size_t)ndst[m] * 64 + j], Sout[t]);
    atomicAdd(&mvagg[(size_t)ndst[m] * 192 + j * 3 + 0], Vout[t * 3 + 0]);
    atomicAdd(&mvagg[(size_t)ndst[m] * 192 + j * 3 + 1], Vout[t * 3 + 1]);
    atomicAdd(&mvagg[(size_t)ndst[m] * 192 + j * 3 + 2], Vout[t * 3 + 2]);
  }
}

// ---------- fused upd1 -> gate -> upd2 -> residual ---------------------------
// 4 waves cooperate on one 16-node tile.
__global__ __launch_bounds__(128) void k_node_upd(
    float* __restrict__ s, float* __restrict__ v,
    const float* __restrict__ msagg, const float* __restrict__ mvagg,
    const float* __restrict__ na0, const float* __restrict__ na1, TPW w1, TPW w2) {
  __shared__ __align__(32) h16 SinF[4 * 512];
  __shared__ __align__(32) h16 SinSF[4 * 512];
  __shared__ h16 VinN[16 * 128 * 3];
  __shared__ __align__(32) h16 tA[4 * 512];
  __shared__ __align__(32) h16 tA2[4 * 512];
  __shared__ float Dso[16 * 128];
  __shared__ float Dsv[16 * 64];
  __shared__ float Vout[16 * 64 * 3];
  __shared__ float Sout[16 * 64];
  __shared__ float a0s[16], a1s[48];
  const int tid = threadIdx.x;
  const int n0 = blockIdx.x * 16;
  if (tid < 16) {
    a0s[tid] = na0[n0 + tid];
    a1s[tid * 3 + 0] = na1[(n0 + tid) * 3 + 0];
    a1s[tid * 3 + 1] = na1[(n0 + tid) * 3 + 1];
    a1s[tid * 3 + 2] = na1[(n0 + tid) * 3 + 2];
  }
  __syncthreads();
  for (int t = tid; t < 16 * 128; t += 128) {
    int m = t >> 7, j = t & 127;
    float val = (j < 64) ? s[(size_t)(n0 + m) * 64 + j]
                         : msagg[(size_t)(n0 + m) * 64 + (j - 64)];
    int fi = fragidx(m, j);
    SinF[fi] = (h16)val;
    SinSF[fi] = (h16)(val * a0s[m]);
  }
  for (int t = tid; t < 16 * 128 * 3; t += 128) {
    int m = t / 384, r = t % 384, i = r / 3, c = r % 3;
    float val = (i < 64) ? v[(size_t)(n0 + m) * 192 + i * 3 + c]
                         : mvagg[(size_t)(n0 + m) * 192 + (i - 64) * 3 + c];
    VinN[t] = (h16)val;
  }
  __syncthreads();
  tp_eval<4, 4, 128, 128, 64>(w1, SinF, SinSF, VinN, a0s, a1s, Dso, Dsv, Vout, tA, tA2);
  tp_gate(64, Dso, 128, Sout, Vout);
  __syncthreads();
  for (int t = tid; t < 1024; t += 128) {
    int m = t >> 6, j = t & 63;
    float val = Sout[t];
    int fi = fragidx(m, j);
    SinF[fi] = (h16)val;
    SinSF[fi] = (h16)(val * a0s[m]);
  }
  for (int t = tid; t < 3072; t += 128) VinN[t] = (h16)Vout[t];
  __syncthreads();
  tp_eval<4, 2, 64, 64, 64>(w2, SinF, SinSF, VinN, a0s, a1s, Dso, Dsv, Vout, tA, tA2);
  for (int t = tid; t < 1024; t += 128) {
    int m = t >> 6, j = t & 63;
    s[(size_t)(n0 + m) * 64 + j] += Dso[t];
    v[(size_t)(n0 + m) * 192 + j * 3 + 0] += Vout[t * 3 + 0];
    v[(size_t)(n0 + m) * 192 + j * 3 + 1] += Vout[t * 3 + 1];
    v[(size_t)(n0 + m) * 192 + j * 3 + 2] += Vout[t * 3 + 2];
  }
}

// ---------------- Instance norm (per graph) ----------------------------------
__global__ void k_in_p1(const float* __restrict__ s, const int* __restrict__ batch,
                        float* __restrict__ gsum, int total) {
  int t = blockIdx.x * blockDim.x + threadIdx.x;
  if (t < total) atomicAdd(&gsum[batch[t >> 6] * 64 + (t & 63)], s[t]);
}

__global__ void k_in_p2(float* __restrict__ s, const float* __restrict__ v,
                        const int* __restrict__ batch, const float* __restrict__ counts,
                        const float* __restrict__ gsum, float* __restrict__ gvar,
                        float* __restrict__ gvn, int total) {
  int t = blockIdx.x * blockDim.x + threadIdx.x;
  if (t >= total) return;
  int n = t >> 6, j = t & 63, g = batch[n];
  float cnt = fmaxf(counts[g], 1.f);
  float sc = s[t] - gsum[g * 64 + j] / cnt;
  s[t] = sc;
  atomicAdd(&gvar[g * 64 + j], sc * sc);
  const float* vp = v + (size_t)n * 192 + j * 3;
  atomicAdd(&gvn[g * 64 + j], vp[0] * vp[0] + vp[1] * vp[1] + vp[2] * vp[2]);
}

__global__ void k_in_p3(float* __restrict__ s, float* __restrict__ v,
                        const int* __restrict__ batch, const float* __restrict__ counts,
                        const float* __restrict__ gvar, const float* __restrict__ gvn,
                        int total) {
  int t = blockIdx.x * blockDim.x + threadIdx.x;
  if (t >= total) return;
  int n = t >> 6, j = t & 63, g = batch[n];
  float cnt = fmaxf(counts[g], 1.f);
  s[t] *= rsqrtf(gvar[g * 64 + j] / cnt + EPSF);
  float rv = rsqrtf(gvn[g * 64 + j] / cnt + EPSF);
  float* vp = v + (size_t)n * 192 + j * 3;
  vp[0] *= rv; vp[1] *= rv; vp[2] *= rv;
}

// ---------------- Attention pooling ------------------------------------------
__device__ void atomicMaxF(float* addr, float val) {
  int* ia = (int*)addr;
  int cur = __float_as_int(*addr);
  while (__int_as_float(cur) < val) {
    int old = atomicCAS(ia, cur, __float_as_int(val));
    if (old == cur) break;
    cur = old;
  }
}

__global__ __launch_bounds__(128) void k_attn_logits(
    const float* __restrict__ s, const int* __restrict__ ntype,
    const int* __restrict__ batch, const float* __restrict__ W1,
    const float* __restrict__ b1, const float* __restrict__ W2,
    const float* __restrict__ b2, float* __restrict__ logits,
    float* __restrict__ gmax) {
  __shared__ float hb[128];
  __shared__ float srow[64];
  const int n = blockIdx.x, j = threadIdx.x;
  if (j < 64) srow[j] = s[(size_t)n * 64 + j];
  __syncthreads();
  float acc = b1[j];
  for (int i = 0; i < 64; ++i) acc += srow[i] * W1[i * 128 + j];
  hb[j] = tanhf(acc);
  __syncthreads();
  if (j < 4) {
    float lg = b2[j];
    for (int i = 0; i < 128; ++i) lg += hb[i] * W2[i * 4 + j];
    float masked = (ntype[n] == 0) ? lg : -1e9f;
    logits[n * 4 + j] = masked;
    atomicMaxF(&gmax[batch[n] * 4 + j], masked);
  }
}

__global__ void k_attn_ex(const float* __restrict__ logits, const int* __restrict__ ntype,
                          const int* __restrict__ batch, const float* __restrict__ gmax,
                          float* __restrict__ exv, float* __restrict__ denom, int total) {
  int t = blockIdx.x * blockDim.x + threadIdx.x;
  if (t >= total) return;
  int n = t >> 2, hd = t & 3, g = batch[n];
  float e = (ntype[n] == 0) ? __expf(logits[t] - gmax[g * 4 + hd]) : 0.f;
  exv[t] = e;
  atomicAdd(&denom[g * 4 + hd], e);
}

__global__ void k_attn_pool(const float* __restrict__ exv, const float* __restrict__ denom,
                            const float* __restrict__ s, const int* __restrict__ batch,
                            float* __restrict__ out, int total) {
  int t = blockIdx.x * blockDim.x + threadIdx.x;
  if (t >= total) return;
  int n = t >> 8, r = t & 255, hd = r >> 6, j = r & 63;
  int g = batch[n];
  float sc = exv[n * 4 + hd] / fmaxf(denom[g * 4 + hd], 1e-16f);
  atomicAdd(&out[g * 256 + hd * 64 + j], sc * s[(size_t)n * 64 + j]);
}

// ---------------- Host launch ------------------------------------------------
extern "C" void kernel_launch(void* const* d_in, const int* in_sizes, int n_in,
                              void* d_out, int out_size, void* d_ws, size_t ws_size,
                              hipStream_t stream) {
  (void)in_sizes; (void)n_in; (void)out_size; (void)ws_size;
  const float* x     = (const float*)d_in[0];
  const float* pos   = (const float*)d_in[1];
  const int*   eidx  = (const int*)d_in[2];
  const int*   batch = (const int*)d_in[3];
  const int*   ntype = (const int*)d_in[4];
  const int*   etype = (const int*)d_in[5];
  const float* W1  = (const float*)d_in[6];
  const float* W2  = (const float*)d_in[7];
  const float* b1  = (const float*)d_in[8];
  const float* b2  = (const float*)d_in[9];
  const float* edge_emb = (const float*)d_in[10];
  const float* ligW   = (const float*)d_in[86];
  const float* ligB   = (const float*)d_in[87];
  const float* ligLB  = (const float*)d_in[88];
  const float* ligG   = (const float*)d_in[89];
  const float* nodeEmb= (const float*)d_in[90];
  const float* protW  = (const float*)d_in[91];
  const float* protB  = (const float*)d_in[92];
  const float* protLB = (const float*)d_in[93];
  const float* protG  = (const float*)d_in[94];

  float* ws = (float*)d_ws;
  size_t o = 0;
  float* s      = ws + o; o += (size_t)NND * 64;
  float* v      = ws + o; o += (size_t)NND * 192;
  float* na0    = ws + o; o += NND;
  float* na1    = ws + o; o += (size_t)NND * 3;
  float* deg    = ws + o; o += NND;
  float* dist   = ws + o; o += NED;
  float* ea1    = ws + o; o += (size_t)NED * 3;
  float* msagg  = ws + o; o += (size_t)NND * 64;
  float* mvagg  = ws + o; o += (size_t)NND * 192;
  float* counts = ws + o; o += NGR;
  float* gsum   = ws + o; o += NGR * 64;
  float* gvar   = ws + o; o += NGR * 64;
  float* gvn    = ws + o; o += NGR * 64;
  float* logits = ws + o; o += (size_t)NND * 4;
  float* exv    = ws + o; o += (size_t)NND * 4;
  float* gmax   = ws + o; o += NGR * 4;
  float* denom  = ws + o; o += NGR * 4;
  o = (o + 7) & ~(size_t)7;  // 32B-align the f16 weight arena
  h16* harena = (h16*)(ws + o);
  size_t ho = 0;

  auto cvt = [&](const float* W, int K, int N) -> const h16* {
    int Kpad = (K + 31) & ~31;
    h16* dst = harena + ho;
    ho += (size_t)Kpad * N;
    int total = Kpad * N;
    k_cvt_w<<<(total + 255) / 256, 256, 0, stream>>>(W, dst, K, N, Kpad);
    return dst;
  };
  auto tpw_at = [&](int base, int sin, int vin, int sout, int vout) {
    TPW t;
    t.Wss   = cvt((const float*)d_in[base + 0], sin, sout);
    t.Wsv   = cvt((const float*)d_in[base + 1], sin, vout);
    t.Wvs   = vin ? cvt((const float*)d_in[base + 2], vin, vout) : nullptr;
    t.Wvv_s = vin ? cvt((const float*)d_in[base + 3], vin, sout) : nullptr;
    t.Wvv_v = vin ? cvt((const float*)d_in[base + 4], vin, vout) : nullptr;
    t.b     = (const float*)d_in[base + 5];
    return t;
  };

  const h16* protWf = cvt(protW, ESMD, 64);
  TPW embed{};
  embed.Wss = cvt((const float*)d_in[11], 64, 64);
  embed.Wsv = cvt((const float*)d_in[12], 64, 64);
  embed.b   = (const float*)d_in[13];
  TPW m1[3], m2[3], u1[3], u2[3];
  for (int l = 0; l < 3; ++l) {
    const int base = 14 + l * 24;
    m1[l] = tpw_at(base,      145, 128, 128, 64);
    m2[l] = tpw_at(base + 6,   64,  64, 128, 64);
    u1[l] = tpw_at(base + 12, 128, 128, 128, 64);
    u2[l] = tpw_at(base + 18,  64,  64,  64, 64);
  }

  auto fill = [&](float* p, float val, size_t n) {
    k_fill<<<(unsigned)((n + 255) / 256), 256, 0, stream>>>(p, val, (int)n);
  };

  fill(deg, 0.f, NND);
  fill(na1, 0.f, (size_t)NND * 3);
  fill(counts, 0.f, NGR);
  k_counts<<<(NND + 255) / 256, 256, 0, stream>>>(batch, counts, NND);
  k_inproj<<<NND / 16, 32, 0, stream>>>(x, ntype, protWf, protB, protG, protLB,
                                        ligW, ligB, ligG, ligLB, nodeEmb, s);
  k_edge_geom<<<(NED + 255) / 256, 256, 0, stream>>>(pos, eidx, dist, ea1, deg, na1, NED);
  k_fin_na<<<(NND + 255) / 256, 256, 0, stream>>>(deg, na0, na1, NND);
  k_embed<<<NND / 16, 32, 0, stream>>>(s, v, na0, na1, embed);

  for (int l = 0; l < 3; ++l) {
    fill(msagg, 0.f, (size_t)NND * 64);
    fill(mvagg, 0.f, (size_t)NND * 192);
    k_edge_msg<<<NED / 16, 128, 0, stream>>>(s, v, dist, ea1, edge_emb, eidx, etype,
                                             m1[l], m2[l], msagg, mvagg, NED);
    k_node_upd<<<NND / 16, 128, 0, stream>>>(s, v, msagg, mvagg, na0, na1, u1[l], u2[l]);
    fill(gsum, 0.f, NGR * 64);
    fill(gvar, 0.f, NGR * 64);
    fill(gvn, 0.f, NGR * 64);
    k_in_p1<<<(NND * 64 + 255) / 256, 256, 0, stream>>>(s, batch, gsum, NND * 64);
    k_in_p2<<<(NND * 64 + 255) / 256, 256, 0, stream>>>(s, v, batch, counts, gsum,
                                                        gvar, gvn, NND * 64);
    k_in_p3<<<(NND * 64 + 255) / 256, 256, 0, stream>>>(s, v, batch, counts, gvar,
                                                        gvn, NND * 64);
  }

  fill(gmax, -1e30f, NGR * 4);
  fill(denom, 0.f, NGR * 4);
  fill((float*)d_out, 0.f, (size_t)NGR * 256);
  k_attn_logits<<<NND, 128, 0, stream>>>(s, ntype, batch, W1, b1, W2, b2, logits, gmax);
  k_attn_ex<<<(NND * 4 + 255) / 256, 256, 0, stream>>>(logits, ntype, batch, gmax,
                                                       exv, denom, NND * 4);
  k_attn_pool<<<(NND * 256 + 255) / 256, 256, 0, stream>>>(exv, denom, s, batch,
                                                           (float*)d_out, NND * 256);
}
